// PytorchHMM_36996848287996
// MI455X (gfx1250) — matrix operand order repaired and checked by reading, not compiled, not run on hardware
//
#include <hip/hip_runtime.h>
#include <hip/hip_bf16.h>

typedef __attribute__((ext_vector_type(16))) _Float16 v16h;
typedef __attribute__((ext_vector_type(8)))  float    v8f;
typedef __attribute__((ext_vector_type(2)))  _Float16 h2t;
typedef __attribute__((ext_vector_type(2)))  __fp16   f16x2;  // return type of cvt_pkrtz

#define B_ 128
#define T_ 4096
#define S_ 64
#define CH 32          // obs chunk length staged in LDS
#define OBS_STRIDE 36  // 16B-aligned row stride, spreads LDS banks

union H2   { _Float16 h[2]; h2t h2; f16x2 f2; unsigned int u; };
union Frag { v16h v; unsigned int u[8]; uint4 q[2]; };

// max over each 16-lane half: xor1/2/4 via DPP8 (VALU), xor8 via ds_swizzle.
// DPP8 selectors in octal (lane7..lane0 3-bit fields), sel[i] = i ^ mask.
__device__ __forceinline__ float halfmax16(float x) {
  float y;
  y = __int_as_float(__builtin_amdgcn_mov_dpp8(__float_as_int(x), 067452301)); // xor 1
  x = fmaxf(x, y);
  y = __int_as_float(__builtin_amdgcn_mov_dpp8(__float_as_int(x), 054761032)); // xor 2
  x = fmaxf(x, y);
  y = __int_as_float(__builtin_amdgcn_mov_dpp8(__float_as_int(x), 032107654)); // xor 4
  x = fmaxf(x, y);
  y = __int_as_float(__builtin_amdgcn_ds_swizzle(__float_as_int(x), 0x201F)); // xor 8 (SWAPX8)
  x = fmaxf(x, y);
  return x;
}

// fast transcendental paths: v_exp_f32 / v_log_f32 (base-2 HW ops) plus one multiply.
__device__ __forceinline__ float fast_exp(float x) {
  return __builtin_amdgcn_exp2f(x * 1.44269504088896341f);   // exp2(x*log2e)
}
__device__ __forceinline__ float fast_log(float x) {
  return __builtin_amdgcn_logf(x) * 0.69314718055994531f;    // v_log_f32 = log2; *ln2
}

__device__ __forceinline__ void load_obs_chunk(const float* __restrict__ obs,
                                               float* s_obs, int b0, int tbase, int lane) {
  if (lane < 16) {
    const float* src = obs + (size_t)(b0 + lane) * T_ + tbase;
    float* dst = s_obs + lane * OBS_STRIDE;
#pragma unroll
    for (int j = 0; j < CH; j += 4) {
      float4 x = *(const float4*)(src + j);
      dst[j] = x.x; dst[j + 1] = x.y; dst[j + 2] = x.z; dst[j + 3] = x.w;
    }
    if (tbase + CH < T_) __builtin_prefetch(src + CH, 0, 0);  // global_prefetch_b8
  }
}

__global__ __launch_bounds__(32) void hmm_forward_wmma(
    const float* __restrict__ obs,   // [B,T]   (C==1)
    const float* __restrict__ tl,    // [S,S]   transition logits
    const float* __restrict__ il,    // [S]     initial logits
    const float* __restrict__ mu,    // [S]     emission means
    const float* __restrict__ elv,   // [S]     emission log-vars
    float* __restrict__ alpha,       // [B,T,S]
    float* __restrict__ ll)          // [B]
{
  __shared__ float    s_max[S_];
  __shared__ float    s_sum[S_];
  __shared__ __align__(16) _Float16 s_w[16 * S_];  // w tile, K-permuted cols [16][64]
  __shared__ float    s_obs[16 * OBS_STRIDE];      // obs chunk [16][CH] padded

  const int lane = threadIdx.x;   // 0..31
  const int n    = lane & 15;     // column within 16x16 tile
  const int hi   = lane >> 4;     // lane-half
  const int b0   = blockIdx.x * 16;

  // ---- transition softmax row stats ----
  for (int r = lane; r < S_; r += 32) {
    float mx = -INFINITY;
    for (int s = 0; s < S_; ++s) mx = fmaxf(mx, tl[r * S_ + s]);
    float sm = 0.f;
    for (int s = 0; s < S_; ++s) sm += expf(tl[r * S_ + s] - mx);
    s_max[r] = mx; s_sum[r] = sm;
  }
  __syncthreads();

  // ---- B fragments: Ptilde[j,s] in f16, WMMA-B layout, with K-dim permutation ----
  // K position k holds source state j(k) = ((k&3)<<4) | (k>>2); pairs are j0, j0+16.
  Frag bfrag[4][2];
#pragma unroll
  for (int nt = 0; nt < 4; ++nt) {
    const int scol = nt * 16 + n;
#pragma unroll
    for (int c = 0; c < 2; ++c) {
#pragma unroll
      for (int v = 0; v < 8; ++v) {
        const int k  = 32 * c + 16 * hi + 2 * v;
        const int j0 = ((k & 3) << 4) | (k >> 2);
        float p0 = expf(tl[j0 * S_ + scol]        - s_max[j0])      / s_sum[j0]      + 1e-10f;
        float p1 = expf(tl[(j0 + 16) * S_ + scol] - s_max[j0 + 16]) / s_sum[j0 + 16] + 1e-10f;
        H2 h; h.h[0] = (_Float16)p0; h.h[1] = (_Float16)p1;
        bfrag[nt][c].u[v] = h.u;
      }
    }
  }

  // ---- per-lane constants: log-init + emission params for this lane's 4 states ----
  float ilmax = -INFINITY;
  for (int s = 0; s < S_; ++s) ilmax = fmaxf(ilmax, il[s]);
  float ilsum = 0.f;
  for (int s = 0; s < S_; ++s) ilsum += expf(il[s] - ilmax);

  float linit[4], e_mu[4], e_c0[4], e_iv[4];
#pragma unroll
  for (int nt = 0; nt < 4; ++nt) {
    const int s = nt * 16 + n;
    linit[nt] = logf(expf(il[s] - ilmax) / ilsum + 1e-10f);
    const float var = expf(elv[s]) + 1e-6f;
    e_mu[nt] = mu[s];
    e_iv[nt] = 0.5f / var;
    e_c0[nt] = -0.5f * logf(6.28318530717958647f * var);
  }

  // ---- t = 0 ----
  load_obs_chunk(obs, s_obs, b0, 0, lane);
  v8f acc[4];  // alpha in WMMA C/D layout: acc[nt][k] = alpha[row k+8*hi, state nt*16+n]
  {
    float ob[8];
#pragma unroll
    for (int k = 0; k < 8; ++k) ob[k] = s_obs[(k + 8 * hi) * OBS_STRIDE + 0];
#pragma unroll
    for (int nt = 0; nt < 4; ++nt) {
#pragma unroll
      for (int k = 0; k < 8; ++k) {
        const float d = ob[k] - e_mu[nt];
        const float a = linit[nt] + e_c0[nt] - d * d * e_iv[nt];
        acc[nt][k] = a;
        alpha[((size_t)(b0 + k + 8 * hi) * T_) * S_ + nt * 16 + n] = a;
      }
    }
  }

  // ---- sequential scan over time ----
  for (int t = 1; t < T_; ++t) {
    const int toff = t & (CH - 1);
    if (toff == 0) load_obs_chunk(obs, s_obs, b0, t, lane);

    // row max over all 64 states: fold 4 tiles, then 16-lane half reduce (DPP8+swizzle)
    float m[8];
#pragma unroll
    for (int k = 0; k < 8; ++k) {
      float mx = fmaxf(fmaxf(acc[0][k], acc[1][k]), fmaxf(acc[2][k], acc[3][k]));
      m[k] = halfmax16(mx);
    }

    // w = exp(alpha - m) -> LDS f16, permuted cols: state nt*16+n -> col n*4+nt.
    // Lane packs its 4 states per row into one 8-byte store.
    uint2* w2 = (uint2*)s_w;
#pragma unroll
    for (int k = 0; k < 8; ++k) {
      const float w0 = fast_exp(acc[0][k] - m[k]);
      const float w1 = fast_exp(acc[1][k] - m[k]);
      const float w2v = fast_exp(acc[2][k] - m[k]);
      const float w3 = fast_exp(acc[3][k] - m[k]);
      H2 pa, pb;
      pa.f2 = __builtin_amdgcn_cvt_pkrtz(w0, w1);
      pb.f2 = __builtin_amdgcn_cvt_pkrtz(w2v, w3);
      w2[(k + 8 * hi) * 16 + n] = make_uint2(pa.u, pb.u);
    }

    // gather A fragments (16-bit A 16x32 layout) as two b128 loads each;
    // same-wave DS ordering => no barrier needed.
    const uint4* wq = (const uint4*)s_w;
    Frag af[2];
#pragma unroll
    for (int c = 0; c < 2; ++c) {
      const int base = n * 8 + 4 * c + 2 * hi;  // uint4 units
      af[c].q[0] = wq[base];
      af[c].q[1] = wq[base + 1];
    }

    float ob[8];
#pragma unroll
    for (int k = 0; k < 8; ++k) ob[k] = s_obs[(k + 8 * hi) * OBS_STRIDE + toff];

#pragma unroll
    for (int nt = 0; nt < 4; ++nt) {
      v8f vv = {};
      vv = __builtin_amdgcn_wmma_f32_16x16x32_f16(false, af[0].v, false, bfrag[nt][0].v,
                                                  (short)0, vv, false, false);
      vv = __builtin_amdgcn_wmma_f32_16x16x32_f16(false, af[1].v, false, bfrag[nt][1].v,
                                                  (short)0, vv, false, false);
#pragma unroll
      for (int k = 0; k < 8; ++k) {
        const float d = ob[k] - e_mu[nt];
        const float a = m[k] + fast_log(vv[k]) + e_c0[nt] - d * d * e_iv[nt];
        acc[nt][k] = a;
        alpha[((size_t)(b0 + k + 8 * hi) * T_ + t) * S_ + nt * 16 + n] = a;
      }
    }
  }

  // ---- log-likelihood: logsumexp over states of alpha[:, T-1, :] ----
  {
    float mf[8];
#pragma unroll
    for (int k = 0; k < 8; ++k) {
      float mx = fmaxf(fmaxf(acc[0][k], acc[1][k]), fmaxf(acc[2][k], acc[3][k]));
      mf[k] = halfmax16(mx);
    }
    float sf[8];
#pragma unroll
    for (int k = 0; k < 8; ++k)
      sf[k] = fast_exp(acc[0][k] - mf[k]) + fast_exp(acc[1][k] - mf[k]) +
              fast_exp(acc[2][k] - mf[k]) + fast_exp(acc[3][k] - mf[k]);
#pragma unroll
    for (int msk = 1; msk <= 8; msk <<= 1) {
#pragma unroll
      for (int k = 0; k < 8; ++k)
        sf[k] += __shfl_xor(sf[k], msk, 32);
    }
    if (n == 0) {
#pragma unroll
      for (int k = 0; k < 8; ++k)
        ll[b0 + k + 8 * hi] = mf[k] + fast_log(sf[k]);
    }
  }
}

extern "C" void kernel_launch(void* const* d_in, const int* in_sizes, int n_in,
                              void* d_out, int out_size, void* d_ws, size_t ws_size,
                              hipStream_t stream) {
  (void)in_sizes; (void)n_in; (void)out_size; (void)d_ws; (void)ws_size;
  const float* obs = (const float*)d_in[0];
  const float* tl  = (const float*)d_in[1];
  const float* il  = (const float*)d_in[2];
  const float* mu  = (const float*)d_in[3];
  const float* elv = (const float*)d_in[4];
  float* alpha = (float*)d_out;
  float* llp   = alpha + (size_t)B_ * T_ * S_;
  hipLaunchKernelGGL(hmm_forward_wmma, dim3(B_ / 16), dim3(32), 0, stream,
                     obs, tl, il, mu, elv, alpha, llp);
}
